// HydraAttention_47734266528139
// MI455X (gfx1250) — matrix operand order repaired and checked by reading, not compiled, not run on hardware
//
#include <hip/hip_runtime.h>
#include <hip/hip_bf16.h>

// ---------------------------------------------------------------------------
// HydraAttention (linear attention) on MI455X / gfx1250, wave32 + WMMA bf16.
//   x[4,4096,1024] @ W_qkv[1024,3072] -> sigmoid(q), tanh(k), v
//   kv = k^T v per (b,h);  out = q @ kv;  y = out @ W_out + b_out
// All GEMMs: v_wmma_f32_16x16x32_bf16, double-buffered LDS, 8-WMMA bursts,
// split load/compute/store software pipeline, CDNA5 async global->LDS copies.
// ---------------------------------------------------------------------------

typedef __attribute__((ext_vector_type(16))) __bf16 v16bf;
typedef __attribute__((ext_vector_type(8)))  float  v8f;
typedef __attribute__((ext_vector_type(4)))  int    v4i;

#define BATCH 4
#define SEQ   4096
#define DIM   1024
#define DIM3  3072
#define MROWS (BATCH * SEQ)   // 16384
#define HD    128             // head dim
#define NCHUNK 8              // sequence chunks for kv partial sums
#define CHUNK (SEQ / NCHUNK)  // 512

// ---- CDNA5 async global->LDS copy (guarded; falls back to plain copy) -----
#if defined(__gfx1250__) && __has_builtin(__builtin_amdgcn_global_load_async_to_lds_b128)
#define USE_ASYNC_LDS 1
#else
#define USE_ASYNC_LDS 0
#endif

#if USE_ASYNC_LDS
typedef __attribute__((address_space(1))) v4i gas_v4i;   // global int4
typedef __attribute__((address_space(3))) v4i las_v4i;   // LDS int4
#endif

__device__ __forceinline__ void copy16_to_lds(unsigned short* dst,
                                              const unsigned short* src) {
#if USE_ASYNC_LDS
    __builtin_amdgcn_global_load_async_to_lds_b128(
        (gas_v4i*)const_cast<unsigned short*>(src),
        (las_v4i*)dst, 0, 0);
#else
    *(uint4*)dst = *(const uint4*)src;
#endif
}

__device__ __forceinline__ void wait_async_copies() {
#if USE_ASYNC_LDS
#if __has_builtin(__builtin_amdgcn_s_wait_asynccnt)
    __builtin_amdgcn_s_wait_asynccnt(0);
#else
    asm volatile("s_wait_asynccnt 0x0" ::: "memory");
#endif
#endif
}

// round-to-nearest-even f32 -> bf16 (raw ushort)
__device__ __forceinline__ unsigned short f32_to_bf16(float f) {
    unsigned int u = __builtin_bit_cast(unsigned int, f);
    u += 0x7FFFu + ((u >> 16) & 1u);
    return (unsigned short)(u >> 16);
}
__device__ __forceinline__ unsigned int pack_bf16x2(float a, float b) {
    return (unsigned int)f32_to_bf16(a) | ((unsigned int)f32_to_bf16(b) << 16);
}

union FragBF { v16bf v; uint4 q[2]; };

// A fragment 16x32 bf16 (this code's contiguous k order): lane = 16*half + m.
// elems 0..15: K = 16*half..+15. As[m][k].
__device__ __forceinline__ void load_a_frag(FragBF& f, const unsigned short* As,
                                            int mBase, int lane) {
    int m = lane & 15, h = lane >> 4;
    const unsigned short* p = As + (mBase + m) * 32;
    f.q[0] = *(const uint4*)(p + 16 * h);
    f.q[1] = *(const uint4*)(p + 16 * h + 8);
}
// B fragment 32x16 bf16: lane = 16*half + n; elems 0..15: K = 16*half..+15.
// LDS staged transposed: Bs[n][k].
__device__ __forceinline__ void load_b_frag(FragBF& f, const unsigned short* Bs,
                                            int nBase, int lane) {
    int n = lane & 15, h = lane >> 4;
    const unsigned short* p = Bs + (nBase + n) * 32 + 16 * h;
    f.q[0] = *(const uint4*)(p);
    f.q[1] = *(const uint4*)(p + 8);
}

// One wave: 16(M) x 128(N) strip; A fragment reused across 8 WMMAs.
__device__ __forceinline__ void mma_strip(v8f acc[8], const unsigned short* As,
                                          const unsigned short* Bs, int wave, int lane) {
    FragBF a;
    load_a_frag(a, As, wave * 16, lane);
    FragBF b[8];
#pragma unroll
    for (int nt = 0; nt < 8; ++nt) load_b_frag(b[nt], Bs, nt * 16, lane);
#pragma unroll
    for (int nt = 0; nt < 8; ++nt)
        acc[nt] = __builtin_amdgcn_wmma_f32_16x16x32_bf16(
            false, a.v, false, b[nt].v, (short)0, acc[nt], false, false);
}

// ---------------------------------------------------------------------------
// Kernel 1: qkv = x @ W_qkv with per-section activation, bf16 out.
// grid = (3072/128, 16384/128), 256 threads (8 waves).
// ---------------------------------------------------------------------------
__global__ __launch_bounds__(256, 1) void qkv_gemm_kernel(
    const float* __restrict__ x, const float* __restrict__ W,
    unsigned short* __restrict__ out) {
    __shared__ __align__(16) unsigned short As[2][128 * 32];
    __shared__ __align__(16) unsigned short Bs[2][128 * 32];
    const int tid = threadIdx.x, wave = tid >> 5, lane = tid & 31;
    const int  colBase = blockIdx.x * 128;
    const long rowBase = (long)blockIdx.y * 128;
    const int  sec = colBase >> 10;           // 0=q(sigmoid) 1=k(tanh) 2=v

    struct St { float4 fa[2][2]; float fb[4][4]; };

    auto stage_load = [&](St& s, int k0) {
#pragma unroll
        for (int it = 0; it < 2; ++it) {      // A: 128x32 f32, 8 elems/thread
            int i = tid + it * 256;
            int r = i >> 2, c = (i & 3) * 8;
            const float* src = &x[(rowBase + r) * DIM + k0 + c];
            s.fa[it][0] = *(const float4*)(src);
            s.fa[it][1] = *(const float4*)(src + 4);
        }
#pragma unroll
        for (int it = 0; it < 4; ++it) {      // B: 4 k-rows per thread
            int i = tid + it * 256;
            int n = i & 127, kk = (i >> 7) * 4;
            const float* src = &W[(long)(k0 + kk) * DIM3 + colBase + n];
#pragma unroll
            for (int j = 0; j < 4; ++j) s.fb[it][j] = src[j * DIM3];
        }
    };
    auto stage_store = [&](const St& s, int buf) {
#pragma unroll
        for (int it = 0; it < 2; ++it) {
            int i = tid + it * 256;
            int r = i >> 2, c = (i & 3) * 8;
            uint4 u;
            u.x = pack_bf16x2(s.fa[it][0].x, s.fa[it][0].y);
            u.y = pack_bf16x2(s.fa[it][0].z, s.fa[it][0].w);
            u.z = pack_bf16x2(s.fa[it][1].x, s.fa[it][1].y);
            u.w = pack_bf16x2(s.fa[it][1].z, s.fa[it][1].w);
            *(uint4*)&As[buf][r * 32 + c] = u;
        }
#pragma unroll
        for (int it = 0; it < 4; ++it) {      // Bs[n][k] transposed
            int i = tid + it * 256;
            int n = i & 127, kk = (i >> 7) * 4;
            uint2 u;
            u.x = pack_bf16x2(s.fb[it][0], s.fb[it][1]);
            u.y = pack_bf16x2(s.fb[it][2], s.fb[it][3]);
            *(uint2*)&Bs[buf][n * 32 + kk] = u;
        }
    };

    v8f acc[8] = {};
    St s;
    stage_load(s, 0);
    stage_store(s, 0);
    int cur = 0;
    for (int k0 = 0; k0 < DIM; k0 += 32) {
        __syncthreads();
        const bool more = (k0 + 32 < DIM);
        if (more) stage_load(s, k0 + 32);     // global loads in flight...
        mma_strip(acc, As[cur], Bs[cur], wave, lane);  // ...behind 8 WMMAs
        if (more) stage_store(s, cur ^ 1);    // loadcnt wait after compute
        cur ^= 1;
    }
    const int half = lane >> 4, nIdx = lane & 15;
#pragma unroll
    for (int nt = 0; nt < 8; ++nt)
#pragma unroll
        for (int r = 0; r < 8; ++r) {
            long row = rowBase + wave * 16 + 8 * half + r;
            int  col = colBase + nt * 16 + nIdx;
            float v = acc[nt][r];
            if (sec == 0)      v = 1.0f / (1.0f + __expf(-v));
            else if (sec == 1) v = tanhf(v);
            out[row * DIM3 + col] = f32_to_bf16(v);
        }
}

// ---------------------------------------------------------------------------
// Kernel 2: partial kv = k^T v over a sequence chunk. grid = (32 bh, 8 chunk).
// ---------------------------------------------------------------------------
__global__ __launch_bounds__(256, 1) void kv_gemm_kernel(
    const unsigned short* __restrict__ qkv, float* __restrict__ part) {
    __shared__ __align__(16) unsigned short As[2][128 * 32];
    __shared__ __align__(16) unsigned short Bs[2][128 * 32];
    const int tid = threadIdx.x, wave = tid >> 5, lane = tid & 31;
    const int bh = blockIdx.x, chunk = blockIdx.y;
    const int b = bh >> 3, h = bh & 7;
    const unsigned short* kb = qkv + (long)b * SEQ * DIM3 + DIM     + h * HD;
    const unsigned short* vb = qkv + (long)b * SEQ * DIM3 + 2 * DIM + h * HD;

    struct St { unsigned short k[2][8], v[2][8]; };

    auto stage_load = [&](St& s, int n0) {
#pragma unroll
        for (int it = 0; it < 2; ++it) {
            int i = tid + it * 256;
            int d = i & 127, nn0 = (i >> 7) * 8;
#pragma unroll
            for (int j = 0; j < 8; ++j) {
                long r = (long)(n0 + nn0 + j) * DIM3;
                s.k[it][j] = kb[r + d];
                s.v[it][j] = vb[r + d];
            }
        }
    };
    auto stage_store = [&](const St& s, int buf) {
#pragma unroll
        for (int it = 0; it < 2; ++it) {
            int i = tid + it * 256;
            int d = i & 127, nn0 = (i >> 7) * 8;
            uint4 ua, ub;
            unsigned int* pa = &ua.x;
            unsigned int* pb = &ub.x;
#pragma unroll
            for (int j = 0; j < 4; ++j) {
                pa[j] = (unsigned int)s.k[it][2 * j] | ((unsigned int)s.k[it][2 * j + 1] << 16);
                pb[j] = (unsigned int)s.v[it][2 * j] | ((unsigned int)s.v[it][2 * j + 1] << 16);
            }
            *(uint4*)&As[buf][d * 32 + nn0] = ua;  // A = k^T : As[d][n]
            *(uint4*)&Bs[buf][d * 32 + nn0] = ub;  // Bs[e][n]
        }
    };

    v8f acc[8] = {};
    const int nStart = chunk * CHUNK;
    St s;
    stage_load(s, nStart);
    stage_store(s, 0);
    int cur = 0;
    for (int n0 = nStart; n0 < nStart + CHUNK; n0 += 32) {
        __syncthreads();
        const bool more = (n0 + 32 < nStart + CHUNK);
        if (more) stage_load(s, n0 + 32);
        mma_strip(acc, As[cur], Bs[cur], wave, lane);
        if (more) stage_store(s, cur ^ 1);
        cur ^= 1;
    }
    float* outp = part + ((long)chunk * 32 + bh) * (HD * HD);
    const int half = lane >> 4, nIdx = lane & 15;
#pragma unroll
    for (int nt = 0; nt < 8; ++nt)
#pragma unroll
        for (int r = 0; r < 8; ++r)
            outp[(wave * 16 + 8 * half + r) * HD + nt * 16 + nIdx] = acc[nt][r];
}

// Deterministic reduction of NCHUNK partials -> bf16 kv.
__global__ __launch_bounds__(256) void kv_reduce_kernel(
    const float* __restrict__ part, unsigned short* __restrict__ kvbf) {
    int idx = blockIdx.x * 256 + threadIdx.x;
    float s = 0.0f;
#pragma unroll
    for (int c = 0; c < NCHUNK; ++c) s += part[(long)c * (32 * HD * HD) + idx];
    kvbf[idx] = f32_to_bf16(s);
}

// ---------------------------------------------------------------------------
// Kernel 3: out = q @ kv per (b,h). grid = (32 bh, 4096/128).
// ---------------------------------------------------------------------------
__global__ __launch_bounds__(256, 1) void attn_out_kernel(
    const unsigned short* __restrict__ qkv, const unsigned short* __restrict__ kv,
    unsigned short* __restrict__ out) {
    __shared__ __align__(16) unsigned short As[2][128 * 32];
    __shared__ __align__(16) unsigned short Bs[2][128 * 32];
    const int tid = threadIdx.x, wave = tid >> 5, lane = tid & 31;
    const int bh = blockIdx.x, b = bh >> 3, h = bh & 7;
    const long m0 = (long)blockIdx.y * 128;
    const unsigned short* qb  = qkv + (long)b * SEQ * DIM3 + h * HD;
    const unsigned short* kvb = kv + (long)bh * (HD * HD);

    struct St { unsigned short b[4][4]; };

    auto stage_load = [&](St& s, int buf, int k0) {
#pragma unroll
        for (int it = 0; it < 2; ++it) {      // A: async 16B copies to LDS
            int i = tid + it * 256;
            int r = i >> 2, c = (i & 3) * 8;
            copy16_to_lds(&As[buf][r * 32 + c], &qb[(m0 + r) * DIM3 + k0 + c]);
        }
#pragma unroll
        for (int it = 0; it < 4; ++it) {      // B: kv rows to regs
            int i = tid + it * 256;
            int e = i & 127, kk = (i >> 7) * 4;
            const unsigned short* src = &kvb[(k0 + kk) * HD + e];
#pragma unroll
            for (int j = 0; j < 4; ++j) s.b[it][j] = src[j * HD];
        }
    };
    auto stage_store = [&](const St& s, int buf) {
#pragma unroll
        for (int it = 0; it < 4; ++it) {      // Bs[e][k] transposed
            int i = tid + it * 256;
            int e = i & 127, kk = (i >> 7) * 4;
            uint2 u;
            u.x = (unsigned int)s.b[it][0] | ((unsigned int)s.b[it][1] << 16);
            u.y = (unsigned int)s.b[it][2] | ((unsigned int)s.b[it][3] << 16);
            *(uint2*)&Bs[buf][e * 32 + kk] = u;
        }
    };

    v8f acc[8] = {};
    St s;
    stage_load(s, 0, 0);
    stage_store(s, 0);
    int cur = 0;
    for (int k0 = 0; k0 < HD; k0 += 32) {
        wait_async_copies();
        __syncthreads();
        const bool more = (k0 + 32 < HD);
        if (more) stage_load(s, cur ^ 1, k0 + 32);
        mma_strip(acc, As[cur], Bs[cur], wave, lane);
        if (more) stage_store(s, cur ^ 1);
        cur ^= 1;
    }
    unsigned short* ob = out + ((long)b * SEQ + m0) * DIM + h * HD;
    const int half = lane >> 4, nIdx = lane & 15;
#pragma unroll
    for (int nt = 0; nt < 8; ++nt)
#pragma unroll
        for (int r = 0; r < 8; ++r)
            ob[(long)(wave * 16 + 8 * half + r) * DIM + nt * 16 + nIdx] =
                f32_to_bf16(acc[nt][r]);
}

// ---------------------------------------------------------------------------
// Kernel 4: y = attn @ W_out + b_out, f32 out. grid = (1024/128, 16384/128).
// ---------------------------------------------------------------------------
__global__ __launch_bounds__(256, 1) void proj_kernel(
    const unsigned short* __restrict__ a, const float* __restrict__ W,
    const float* __restrict__ bias, float* __restrict__ y) {
    __shared__ __align__(16) unsigned short As[2][128 * 32];
    __shared__ __align__(16) unsigned short Bs[2][128 * 32];
    const int tid = threadIdx.x, wave = tid >> 5, lane = tid & 31;
    const int  colBase = blockIdx.x * 128;
    const long rowBase = (long)blockIdx.y * 128;

    struct St { float b[4][4]; };

    auto stage_load = [&](St& s, int buf, int k0) {
#pragma unroll
        for (int it = 0; it < 2; ++it) {      // A: async 16B copies to LDS
            int i = tid + it * 256;
            int r = i >> 2, c = (i & 3) * 8;
            copy16_to_lds(&As[buf][r * 32 + c], &a[(rowBase + r) * DIM + k0 + c]);
        }
#pragma unroll
        for (int it = 0; it < 4; ++it) {      // B: W_out f32 rows to regs
            int i = tid + it * 256;
            int n = i & 127, kk = (i >> 7) * 4;
            const float* src = &W[(long)(k0 + kk) * DIM + colBase + n];
#pragma unroll
            for (int j = 0; j < 4; ++j) s.b[it][j] = src[j * DIM];
        }
    };
    auto stage_store = [&](const St& s, int buf) {
#pragma unroll
        for (int it = 0; it < 4; ++it) {      // Bs[n][k] transposed, bf16
            int i = tid + it * 256;
            int n = i & 127, kk = (i >> 7) * 4;
            uint2 u;
            u.x = pack_bf16x2(s.b[it][0], s.b[it][1]);
            u.y = pack_bf16x2(s.b[it][2], s.b[it][3]);
            *(uint2*)&Bs[buf][n * 32 + kk] = u;
        }
    };

    v8f acc[8] = {};
    St s;
    stage_load(s, 0, 0);
    stage_store(s, 0);
    int cur = 0;
    for (int k0 = 0; k0 < DIM; k0 += 32) {
        wait_async_copies();
        __syncthreads();
        const bool more = (k0 + 32 < DIM);
        if (more) stage_load(s, cur ^ 1, k0 + 32);
        mma_strip(acc, As[cur], Bs[cur], wave, lane);
        if (more) stage_store(s, cur ^ 1);
        cur ^= 1;
    }
    const int half = lane >> 4, nIdx = lane & 15;
#pragma unroll
    for (int nt = 0; nt < 8; ++nt)
#pragma unroll
        for (int r = 0; r < 8; ++r) {
            long row = rowBase + wave * 16 + 8 * half + r;
            int  col = colBase + nt * 16 + nIdx;
            y[row * DIM + col] = acc[nt][r] + bias[col];
        }
}

// ---------------------------------------------------------------------------
extern "C" void kernel_launch(void* const* d_in, const int* in_sizes, int n_in,
                              void* d_out, int out_size, void* d_ws, size_t ws_size,
                              hipStream_t stream) {
    const float* x    = (const float*)d_in[0];
    const float* Wqkv = (const float*)d_in[1];
    const float* Wout = (const float*)d_in[2];
    const float* bout = (const float*)d_in[3];
    float*       y    = (float*)d_out;

    // workspace layout (~152 MB):
    unsigned short* qkv_bf  = (unsigned short*)d_ws;                   // 16384*3072 bf16
    unsigned short* attn_bf = qkv_bf + (size_t)MROWS * DIM3;           // 16384*1024 bf16
    unsigned short* kv_bf   = attn_bf + (size_t)MROWS * DIM;           // 32*128*128 bf16
    float*          kv_part = (float*)(kv_bf + (size_t)32 * HD * HD);  // 8*32*128*128 f32

    qkv_gemm_kernel<<<dim3(DIM3 / 128, MROWS / 128), 256, 0, stream>>>(x, Wqkv, qkv_bf);
    kv_gemm_kernel<<<dim3(32, NCHUNK), 256, 0, stream>>>(qkv_bf, kv_part);
    kv_reduce_kernel<<<dim3((32 * HD * HD) / 256), 256, 0, stream>>>(kv_part, kv_bf);
    attn_out_kernel<<<dim3(32, SEQ / 128), 256, 0, stream>>>(qkv_bf, kv_bf, attn_bf);
    proj_kernel<<<dim3(DIM / 128, MROWS / 128), 256, 0, stream>>>(attn_bf, Wout, bout, y);
}
